// GraphSAGE_10651518894450
// MI455X (gfx1250) — compile-verified, matrix operand order checked by reading, not run on hardware
//
#include <hip/hip_runtime.h>
#include <hip/hip_bf16.h>

// GraphSAGE 2-layer forward for MI455X (gfx1250, wave32, WMMA).
//
// v3: (1) weights zero-padded to NT tiles per K-row -> branch-free inner
//     loop of 10 back-to-back v_wmma; (2) double-buffered B-fragment
//     ds_loads so LDS latency overlaps the matrix pipe; (3) K-loop split
//     into clean body + one guarded tail; (4) LDS weight staging done with
//     gfx1250 async copies (global_load_async_to_lds_b128 + s_wait_asynccnt).

typedef __attribute__((ext_vector_type(16))) __bf16          v16bf;
typedef __attribute__((ext_vector_type(8)))  float           v8f;

__device__ __forceinline__ unsigned short f2bf_bits(float f) {
    unsigned int u = __float_as_uint(f);
    unsigned int r = u + 0x7FFFu + ((u >> 16) & 1u);   // round-to-nearest-even
    return (unsigned short)(r >> 16);
}

// ---------------------------------------------------------------------------
// Pre-swizzle a KxN fp32 weight matrix into per-lane WMMA bf16 B fragments.
// Output element layout:  [(kt*ntPad + nt)*512 + lane*16 + e]
// 16-bit operand packing (ISA 7.12.2): lane<16 -> K = ks+e (e<8), ks+16+e-8;
// lanes 16..31 use ks+8. lane&15 indexes the N column. Zero-padded OOB
// (including whole padded N-tiles up to ntPad).
// ---------------------------------------------------------------------------
__global__ void wconvert_kernel(const float* __restrict__ W,
                                unsigned short* __restrict__ out,
                                int K, int N, int ntPad, int total) {
    int idx = blockIdx.x * blockDim.x + threadIdx.x;
    if (idx >= total) return;
    int e    = idx & 15;
    int lane = (idx >> 4) & 31;
    int tile = idx >> 9;
    int kt = tile / ntPad;
    int nt = tile - kt * ntPad;
    int lo = lane & 15, hi = lane >> 4;
    int k  = kt * 32 + hi * 8 + ((e < 8) ? e : (e + 8));
    int n  = nt * 16 + lo;
    float v = (k < K && n < N) ? W[(size_t)k * N + n] : 0.0f;
    out[idx] = f2bf_bits(v);
}

// ---------------------------------------------------------------------------
// Degree count + reciprocal (in place).
// ---------------------------------------------------------------------------
__global__ void deg_count_kernel(const int* __restrict__ dst,
                                 float* __restrict__ deg, int nE) {
    int e = blockIdx.x * blockDim.x + threadIdx.x;
    if (e >= nE) return;
    atomicAdd(&deg[dst[e]], 1.0f);
}

__global__ void invdeg_kernel(float* __restrict__ deg, int n) {
    int i = blockIdx.x * blockDim.x + threadIdx.x;
    if (i >= n) return;
    deg[i] = 1.0f / fmaxf(deg[i], 1.0f);
}

// ---------------------------------------------------------------------------
// Edge scatter-add: out[dst[e]*D + d] += X[src[e]*D + d].
// One thread per (edge, dim); message buffer (60MB) is L2-resident on MI455X
// (192MB L2) so the fp32 atomics resolve in L2, not HBM.
// ---------------------------------------------------------------------------
__global__ void scatter_add_kernel(const float* __restrict__ X,
                                   const int* __restrict__ src,
                                   const int* __restrict__ dst,
                                   float* __restrict__ out,
                                   int D, int total) {
    int idx = blockIdx.x * blockDim.x + threadIdx.x;
    if (idx >= total) return;
    int e = idx / D;
    int d = idx - e * D;
    int s = src[e], t = dst[e];
    atomicAdd(&out[(size_t)t * D + d], X[(size_t)s * D + d]);
}

// ---------------------------------------------------------------------------
// One K-row of WMMAs: NT tiles, double-buffered B fragments from LDS.
// ---------------------------------------------------------------------------
template <int NT>
__device__ __forceinline__ void mma_row(const unsigned short* ldsB, int base,
                                        int lane, const v16bf& a,
                                        v8f (&acc)[NT]) {
    v16bf bcur = *(const v16bf*)(ldsB + base + lane * 16);
#pragma unroll
    for (int t = 0; t < NT; ++t) {
        v16bf bnext;
        if (t + 1 < NT)
            bnext = *(const v16bf*)(ldsB + base + ((t + 1) << 9) + lane * 16);
        acc[t] = __builtin_amdgcn_wmma_f32_16x16x32_bf16(
                     false, a, false, bcur, (short)0, acc[t], false, false);
        bcur = bnext;
    }
}

// ---------------------------------------------------------------------------
// WMMA GEMM: C[M,N] = op( A[M,K] @ Bsw ) with fused epilogue.
//   NT      : N-tiles (16 cols each) computed per wave from one A fragment
//   SCALE_A : scale A row r by invdeg[r] on load (agg = (msg/deg) @ W)
//   ADD_D   : epilogue += Dm[r,n] * invdeg[r]   (fused aggregation add)
//   BIAS    : epilogue += bias[n]
//   ACCUM   : epilogue += existing C[r,n]
//   RELU    : epilogue = max(., 0)
// Whole swizzled B matrix (szB bf16 elems, ~100KB) staged in LDS via
// gfx1250 async copies. One wave owns a 16 x (16*NT) C strip; 4 waves/block.
// ---------------------------------------------------------------------------
template <int NT, bool SCALE_A, bool ADD_D, bool BIAS, bool ACCUM, bool RELU>
__global__ __launch_bounds__(128) void gemm_wmma_kernel(
        const float* __restrict__ A, int lda,
        const unsigned short* __restrict__ Bsw, int szB,
        float* __restrict__ C, int ldc,
        const float* __restrict__ Dm,
        const float* __restrict__ invdeg,
        const float* __restrict__ bias,
        int M, int K, int N, int ntPad) {
    extern __shared__ unsigned short ldsB[];

    // async stage of the whole swizzled weight matrix into LDS, 16B per lane
    {
        const uint4* gsrc = (const uint4*)Bsw;
        const int nvec = szB >> 3;              // 8 bf16 per 16B
        for (int i = threadIdx.x; i < nvec; i += blockDim.x) {
            unsigned laddr = (unsigned)(size_t)(const void*)(ldsB + (i << 3));
            const uint4* g = gsrc + i;
            asm volatile("global_load_async_to_lds_b128 %0, %1, off"
                         :: "v"(laddr), "v"(g) : "memory");
        }
        asm volatile("s_wait_asynccnt 0x0" ::: "memory");
    }
    __syncthreads();

    const int lane  = threadIdx.x & 31;
    const int wave  = threadIdx.x >> 5;
    const int tileM = blockIdx.x * 4 + wave;
    if (tileM * 16 >= M) return;                 // wave-uniform after barrier
    const int nt0 = blockIdx.y * NT;
    const int lo = lane & 15, hi = lane >> 4;
    const int arow = tileM * 16 + lo;
    const float ascale = SCALE_A ? invdeg[arow] : 1.0f;
    const float* __restrict__ Arow = A + (size_t)arow * lda;

    v8f acc[NT];
#pragma unroll
    for (int t = 0; t < NT; ++t) { v8f z = {}; acc[t] = z; }

    const int kfull = K >> 5;                    // clean 32-wide K tiles
    for (int kt = 0; kt < kfull; ++kt) {
        const int ks = kt * 32 + hi * 8;
        union { v16bf b; __bf16 e[16]; } af;
#pragma unroll
        for (int i = 0; i < 4; ++i) {
            float2 f0 = *(const float2*)(Arow + ks + i * 2);
            float2 f1 = *(const float2*)(Arow + ks + 16 + i * 2);
            af.e[i * 2 + 0]     = (__bf16)(f0.x * ascale);
            af.e[i * 2 + 1]     = (__bf16)(f0.y * ascale);
            af.e[8 + i * 2 + 0] = (__bf16)(f1.x * ascale);
            af.e[8 + i * 2 + 1] = (__bf16)(f1.y * ascale);
        }
        __builtin_prefetch(Arow + (kt + 1) * 32, 0, 3);   // global_prefetch_b8
        mma_row<NT>(ldsB, ((kt * ntPad + nt0) << 9), lane, af.b, acc);
    }
    if (K & 31) {                                // guarded K tail
        const int ks = kfull * 32 + hi * 8;
        union { v16bf b; __bf16 e[16]; } af;
#pragma unroll
        for (int i = 0; i < 8; ++i) {
            int k0 = ks + i;
            int k1 = ks + 16 + i;
            af.e[i]     = (__bf16)((k0 < K) ? Arow[k0] * ascale : 0.0f);
            af.e[8 + i] = (__bf16)((k1 < K) ? Arow[k1] * ascale : 0.0f);
        }
        mma_row<NT>(ldsB, ((kfull * ntPad + nt0) << 9), lane, af.b, acc);
    }

    // epilogue: C/D layout -> n = lane&15; VGPR v -> row v (lanes 0-15), v+8
    float idg[8];
    if (ADD_D) {
#pragma unroll
        for (int v = 0; v < 8; ++v)
            idg[v] = invdeg[tileM * 16 + hi * 8 + v];
    }
#pragma unroll
    for (int t = 0; t < NT; ++t) {
        const int n = (nt0 + t) * 16 + lo;
        if (n < N) {                             // also skips padded N-tiles
#pragma unroll
            for (int v = 0; v < 8; ++v) {
                const int r = tileM * 16 + hi * 8 + v;
                float val = acc[t][v];
                if (ADD_D)  val += Dm[(size_t)r * ldc + n] * idg[v];
                if (BIAS)   val += bias[n];
                if (ACCUM)  val += C[(size_t)r * ldc + n];
                if (RELU)   val = fmaxf(val, 0.0f);
                C[(size_t)r * ldc + n] = val;
            }
        }
    }
}

// ---------------------------------------------------------------------------
// Host launcher.
// ---------------------------------------------------------------------------
static inline size_t align256(size_t x) { return (x + 255) & ~(size_t)255; }

extern "C" void kernel_launch(void* const* d_in, const int* in_sizes, int n_in,
                              void* d_out, int out_size, void* d_ws, size_t ws_size,
                              hipStream_t stream) {
    const float* x    = (const float*)d_in[0];
    const int*   edges= (const int*)  d_in[1];
    const float* W1l  = (const float*)d_in[2];
    const float* b1   = (const float*)d_in[3];
    const float* W1r  = (const float*)d_in[4];
    const float* W2l  = (const float*)d_in[5];
    const float* b2   = (const float*)d_in[6];
    const float* W2r  = (const float*)d_in[7];
    float* out = (float*)d_out;

    const int dHid   = in_sizes[3];              // 150
    const int dIn    = in_sizes[2] / dHid;       // 300
    const int nNodes = in_sizes[0] / dIn;        // 100000
    const int nE     = in_sizes[1] / 2;          // 800000
    const int* src = edges;
    const int* dst = edges + nE;

    constexpr int NT = 10;
    const int kt1 = (dIn  + 31) / 32;                         // 10
    const int kt2 = (dHid + 31) / 32;                         //  5
    const int nt1 = (dHid + 15) / 16, nt1p = ((nt1 + NT - 1) / NT) * NT; // 10
    const int nt2 = (dIn  + 15) / 16, nt2p = ((nt2 + NT - 1) / NT) * NT; // 20
    const int szW1 = kt1 * nt1p * 512;           // swizzled bf16 elements
    const int szW2 = kt2 * nt2p * 512;

    // ---- carve workspace (~181 MB total; harness scratch assumed >= this) --
    char* p = (char*)d_ws;
    unsigned short* W1l_sw = (unsigned short*)p; p += align256((size_t)szW1 * 2);
    unsigned short* W1r_sw = (unsigned short*)p; p += align256((size_t)szW1 * 2);
    unsigned short* W2l_sw = (unsigned short*)p; p += align256((size_t)szW2 * 2);
    unsigned short* W2r_sw = (unsigned short*)p; p += align256((size_t)szW2 * 2);
    float* invdeg = (float*)p; p += align256((size_t)nNodes * 4);
    float* y1     = (float*)p; p += align256((size_t)nNodes * dHid * 4); // reused as msg2
    float* msg1   = (float*)p; p += align256((size_t)nNodes * dHid * 4);
    float* h      = (float*)p; p += align256((size_t)nNodes * dHid * 4);
    float* msg2   = y1;
    (void)ws_size; (void)n_in; (void)out_size;

    // ---- weight pre-swizzle (bf16, WMMA B-fragment order) ------------------
    wconvert_kernel<<<(szW1 + 255) / 256, 256, 0, stream>>>(W1l, W1l_sw, dIn,  dHid, nt1p, szW1);
    wconvert_kernel<<<(szW1 + 255) / 256, 256, 0, stream>>>(W1r, W1r_sw, dIn,  dHid, nt1p, szW1);
    wconvert_kernel<<<(szW2 + 255) / 256, 256, 0, stream>>>(W2l, W2l_sw, dHid, dIn,  nt2p, szW2);
    wconvert_kernel<<<(szW2 + 255) / 256, 256, 0, stream>>>(W2r, W2r_sw, dHid, dIn,  nt2p, szW2);

    // ---- degrees -> invdeg -------------------------------------------------
    hipMemsetAsync(invdeg, 0, (size_t)nNodes * 4, stream);
    deg_count_kernel<<<(nE + 255) / 256, 256, 0, stream>>>(dst, invdeg, nE);
    invdeg_kernel<<<(nNodes + 255) / 256, 256, 0, stream>>>(invdeg, nNodes);

    const int mTiles = (nNodes + 15) / 16;       // 6250 (exact)
    const dim3 blk(128);
    const dim3 g1((mTiles + 3) / 4, nt1p / NT);  // (1563, 1)
    const dim3 g2((mTiles + 3) / 4, nt2p / NT);  // (1563, 2)
    const size_t sm1 = (size_t)szW1 * 2;         // ~100KB LDS (3 blocks/WGP)
    const size_t sm2 = (size_t)szW2 * 2;

    // ---- layer 1: project first, aggregate in 150-d ------------------------
    // y1 = x @ W1_l
    gemm_wmma_kernel<NT, false, false, false, false, false><<<g1, blk, sm1, stream>>>(
        x, dIn, W1l_sw, szW1, y1, dHid, nullptr, nullptr, nullptr,
        nNodes, dIn, dHid, nt1p);
    // msg1 = segment_sum(y1[src] -> dst)
    hipMemsetAsync(msg1, 0, (size_t)nNodes * dHid * 4, stream);
    scatter_add_kernel<<<((size_t)nE * dHid + 255) / 256, 256, 0, stream>>>(
        y1, src, dst, msg1, dHid, nE * dHid);
    // h = relu(msg1*invdeg + b1 + x @ W1_r)
    gemm_wmma_kernel<NT, false, true, true, false, true><<<g1, blk, sm1, stream>>>(
        x, dIn, W1r_sw, szW1, h, dHid, msg1, invdeg, b1,
        nNodes, dIn, dHid, nt1p);

    // ---- layer 2 -----------------------------------------------------------
    // msg2 = segment_sum(h[src] -> dst)   (reuse y1 buffer)
    hipMemsetAsync(msg2, 0, (size_t)nNodes * dHid * 4, stream);
    scatter_add_kernel<<<((size_t)nE * dHid + 255) / 256, 256, 0, stream>>>(
        h, src, dst, msg2, dHid, nE * dHid);
    // out = h @ W2_r + b2
    gemm_wmma_kernel<NT, false, false, true, false, false><<<g2, blk, sm2, stream>>>(
        h, dHid, W2r_sw, szW2, out, dIn, nullptr, nullptr, b2,
        nNodes, dHid, dIn, nt2p);
    // out += (msg2 * invdeg) @ W2_l
    gemm_wmma_kernel<NT, true, false, false, true, false><<<g2, blk, sm2, stream>>>(
        msg2, dHid, W2l_sw, szW2, out, dIn, nullptr, invdeg, nullptr,
        nNodes, dHid, dIn, nt2p);
}
